// RelationTransformerLayer_75471165325849
// MI455X (gfx1250) — matrix operand order repaired and checked by reading, not compiled, not run on hardware
//
#include <hip/hip_runtime.h>
#include <hip/hip_bf16.h>
#include <stdint.h>

// ---------------------------------------------------------------------------
// Problem dims (fixed by the reference)
// ---------------------------------------------------------------------------
constexpr int Bc   = 4;
constexpr int Nn   = 128;
constexpr int Dc   = 256;
constexpr int Hc   = 8;
constexpr int DKc  = 32;
constexpr int DFFc = 1024;
constexpr int BN   = Bc * Nn;                        // 512
constexpr long long BNN = (long long)Bc * Nn * Nn;   // 65536

// ---------------------------------------------------------------------------
// WMMA tile primitive: 16x16 f32 tile, K-chunks of 32, bf16 inputs
// ---------------------------------------------------------------------------
typedef __attribute__((ext_vector_type(16))) __bf16 v16bf;
typedef __attribute__((ext_vector_type(8)))  float  v8f;

__device__ __forceinline__ v8f wmma_bf16(v8f c, v16bf a, v16bf b) {
  return __builtin_amdgcn_wmma_f32_16x16x32_bf16(false, a, false, b,
                                                 (short)0, c, false, false);
}

// A (16x32, bf16): lane l holds row l%16; half=l/16 selects K sub-blocks (ISA 7.12.2)
template <class F>
__device__ __forceinline__ v16bf frag_a(F f) {
  const int l = threadIdx.x & 31;
  const int row = l & 15, half = l >> 4;
  v16bf a;
#pragma unroll
  for (int e = 0; e < 16; ++e) {
    const int k = half * 16 + e;
    a[e] = (__bf16)f(row, k);
  }
  return a;
}

// B (32x16, bf16): lane l holds col l%16; half=l/16 selects K 16-chunk.
// With [N][K] bf16 weight layout each lane reads 32 contiguous bytes.
template <class F>
__device__ __forceinline__ v16bf frag_b(F f) {
  const int l = threadIdx.x & 31;
  const int col = l & 15, half = l >> 4;
  v16bf b;
#pragma unroll
  for (int e = 0; e < 16; ++e) b[e] = (__bf16)f(half * 16 + e, col);
  return b;
}

// C/D (16x16, f32): VGPR r holds row r + 8*(lane/16), col lane%16
template <class F>
__device__ __forceinline__ void tile_store(const v8f& c, F f) {
  const int l = threadIdx.x & 31;
  const int col = l & 15, half = l >> 4;
#pragma unroll
  for (int r = 0; r < 8; ++r) f(r + 8 * half, col, c[r]);
}

// ---------------------------------------------------------------------------
// Weight prep: W[K x N] f32 (row-major)  ->  Wt[N x K] bf16 (row-major)
// 16x16 LDS tile transpose; both global sides coalesced. Runs once per launch.
// ---------------------------------------------------------------------------
__global__ void k_wt_bf16(const float* __restrict__ W, __bf16* __restrict__ Wt,
                          int K, int N) {
  __shared__ float tile[16][17];
  const int kt = blockIdx.x, nt = blockIdx.y;
  const int tx = threadIdx.x & 15, ty = threadIdx.x >> 4;
  tile[ty][tx] = W[(size_t)(kt * 16 + ty) * N + nt * 16 + tx];
  __syncthreads();
  Wt[(size_t)(nt * 16 + ty) * K + kt * 16 + tx] = (__bf16)tile[tx][ty];
}

// ---------------------------------------------------------------------------
// Generic GEMM:  C[M x Nc] = act(A[M x K] @ Wt^T + bias),  Wt is [Nc x K] bf16
// one wave per 16x16 tile; 4 waves / block
// ---------------------------------------------------------------------------
__global__ void k_gemm_bias(const float* __restrict__ A, int lda,
                            const __bf16* __restrict__ Wt,
                            const float* __restrict__ bias,
                            float* __restrict__ C, int ldc,
                            int M, int Ncols, int K, int do_relu) {
  const int mt = M >> 4, nt = Ncols >> 4;
  const int tile = blockIdx.x * (blockDim.x >> 5) + (threadIdx.x >> 5);
  if (tile >= mt * nt) return;                    // wave-uniform guard
  const int mi = tile / nt, ni = tile % nt;
  const float*  Ab = A  + (size_t)mi * 16 * lda;
  const __bf16* Wb = Wt + (size_t)ni * 16 * K;
  v8f acc = {};
  for (int kb = 0; kb < K; kb += 32) {
    v16bf af = frag_a([&](int r, int k) { return Ab[(size_t)r * lda + kb + k]; });
    v16bf bf = frag_b([&](int k, int c) { return Wb[(size_t)c * K + kb + k]; });
    acc = wmma_bf16(acc, af, bf);
  }
  tile_store(acc, [&](int r, int c, float v) {
    float val = v + bias[ni * 16 + c];
    if (do_relu) val = val > 0.f ? val : 0.f;
    C[(size_t)(mi * 16 + r) * ldc + ni * 16 + c] = val;
  });
}

// ---------------------------------------------------------------------------
// R[b,x,h,c] = sum_d Wrk[c, h*32+d] * q[b,x, h*32+d]     (K = 32, one chunk)
// Wrk accessed as [c][d] row-major -> already contiguous in the contraction dim
// ---------------------------------------------------------------------------
__global__ void k_relR(const float* __restrict__ q, const float* __restrict__ Wrk,
                       float* __restrict__ R) {
  const int ni = blockIdx.x;   // 0..15  (c tiles)
  const int mi = blockIdx.y;   // 0..31  (b*n row tiles)
  const int h  = blockIdx.z;   // 0..7
  v8f acc = {};
  v16bf af = frag_a([&](int r, int k) {
    return q[(size_t)(mi * 16 + r) * Dc + h * DKc + k];
  });
  v16bf bf = frag_b([&](int k, int c) {
    return Wrk[(size_t)(ni * 16 + c) * Dc + h * DKc + k];
  });
  acc = wmma_bf16(acc, af, bf);
  tile_store(acc, [&](int r, int c, float v) {
    R[((size_t)(mi * 16 + r) * Hc + h) * Dc + ni * 16 + c] = v;
  });
}

// ---------------------------------------------------------------------------
// scores:  S[b,x,y,h] = (graphs[b,x,y,:].R[b,x,h,:] + k[b,y,:].q_blockdiag)/sqrt(32)
// one wave: 16 y-rows x 16 cols (h padded 8->16). grid (ytile=8, bx=512)
// ---------------------------------------------------------------------------
__global__ void k_scores(const float* __restrict__ graphs,
                         const float* __restrict__ R,
                         const float* __restrict__ qf,
                         const float* __restrict__ kf,
                         const uint8_t* __restrict__ mask,
                         float* __restrict__ S) {
  const int yt = blockIdx.x;       // 0..7
  const int bx = blockIdx.y;       // 0..511 (= b*128 + x)
  const int b  = bx >> 7;
  v8f acc = {};
  // term 1: graphs . R   (contraction over c = 256; R contiguous in c)
  for (int kb = 0; kb < Dc; kb += 32) {
    v16bf af = frag_a([&](int r, int k) {
      return graphs[((size_t)bx * Nn + yt * 16 + r) * Dc + kb + k];
    });
    v16bf bf = frag_b([&](int k, int c) {
      return (c < Hc) ? R[((size_t)bx * Hc + c) * Dc + kb + k] : 0.f;
    });
    acc = wmma_bf16(acc, af, bf);
  }
  // term 2: k . q  (block-diagonal per head; contraction over c = 256)
  for (int kb = 0; kb < Dc; kb += 32) {
    v16bf af = frag_a([&](int r, int k) {
      return kf[((size_t)b * Nn + yt * 16 + r) * Dc + kb + k];
    });
    v16bf bf = frag_b([&](int k, int c) {
      const int cc = kb + k;
      return (c < Hc && (cc >> 5) == c) ? qf[(size_t)bx * Dc + cc] : 0.f;
    });
    acc = wmma_bf16(acc, af, bf);
  }
  tile_store(acc, [&](int r, int c, float v) {
    if (c >= Hc) return;
    const int y = yt * 16 + r;
    float s = v * 0.17677669529663689f;          // 1/sqrt(32)
    if (mask[(size_t)bx * Nn + y]) s = -1e9f;
    S[((size_t)bx * Nn + y) * Hc + c] = s;
  });
}

// ---------------------------------------------------------------------------
// softmax over y (in place on S). grid = B*N*H rows, 128 threads = 128 y
// ---------------------------------------------------------------------------
__global__ void k_softmax(float* __restrict__ S) {
  __shared__ float sh[Nn];
  const int rid = blockIdx.x;              // 0..4095
  const int bx = rid >> 3, h = rid & 7;
  const int y = threadIdx.x;
  const size_t idx = ((size_t)bx * Nn + y) * Hc + h;
  const float v = S[idx];
  sh[y] = v; __syncthreads();
  for (int o = 64; o > 0; o >>= 1) { if (y < o) sh[y] = fmaxf(sh[y], sh[y + o]); __syncthreads(); }
  const float mx = sh[0]; __syncthreads();
  const float e = __expf(v - mx);
  sh[y] = e; __syncthreads();
  for (int o = 64; o > 0; o >>= 1) { if (y < o) sh[y] += sh[y + o]; __syncthreads(); }
  S[idx] = e / sh[0];
}

// ---------------------------------------------------------------------------
// t[b,x,h,c] = sum_y p[b,x,y,h] * graphs[b,x,y,c]   (M=8 padded to 16, K=128)
// grid (ct=16, bx=512)
// ---------------------------------------------------------------------------
__global__ void k_t(const float* __restrict__ P, const float* __restrict__ graphs,
                    float* __restrict__ T) {
  const int ni = blockIdx.x, bx = blockIdx.y;
  v8f acc = {};
  for (int kb = 0; kb < Nn; kb += 32) {
    v16bf af = frag_a([&](int r, int k) {
      return (r < Hc) ? P[((size_t)bx * Nn + kb + k) * Hc + r] : 0.f;
    });
    v16bf bf = frag_b([&](int k, int c) {
      return graphs[((size_t)bx * Nn + kb + k) * Dc + ni * 16 + c];
    });
    acc = wmma_bf16(acc, af, bf);
  }
  tile_store(acc, [&](int r, int c, float v) {
    if (r < Hc) T[((size_t)bx * Hc + r) * Dc + ni * 16 + c] = v;
  });
}

// ---------------------------------------------------------------------------
// attn[b,x,h*32+d] = sum_y p*v  +  sum_c t*Wrv  + brv    grid (nt=2, mt=8, b*8+h)
// Wrvt is [256 x 256] bf16, [out][in] layout
// ---------------------------------------------------------------------------
__global__ void k_attn(const float* __restrict__ P, const float* __restrict__ vf,
                       const float* __restrict__ T, const __bf16* __restrict__ Wrvt,
                       const float* __restrict__ brv, float* __restrict__ attn) {
  const int ni = blockIdx.x, mi = blockIdx.y;
  const int b = blockIdx.z >> 3, h = blockIdx.z & 7;
  v8f acc = {};
  // p @ v_head  (K = 128 over y)
  for (int kb = 0; kb < Nn; kb += 32) {
    v16bf af = frag_a([&](int r, int k) {
      return P[(((size_t)b * Nn + mi * 16 + r) * Nn + kb + k) * Hc + h];
    });
    v16bf bf = frag_b([&](int k, int c) {
      return vf[((size_t)b * Nn + kb + k) * Dc + h * DKc + ni * 16 + c];
    });
    acc = wmma_bf16(acc, af, bf);
  }
  // t_head @ Wrv_head  (K = 256 over c; Wrvt contiguous in c)
  for (int kb = 0; kb < Dc; kb += 32) {
    v16bf af = frag_a([&](int r, int k) {
      return T[(((size_t)b * Nn + mi * 16 + r) * Hc + h) * Dc + kb + k];
    });
    v16bf bf = frag_b([&](int k, int c) {
      return Wrvt[(size_t)(h * DKc + ni * 16 + c) * Dc + kb + k];
    });
    acc = wmma_bf16(acc, af, bf);
  }
  tile_store(acc, [&](int r, int c, float v) {
    attn[((size_t)b * Nn + mi * 16 + r) * Dc + h * DKc + ni * 16 + c] =
        v + brv[h * DKc + ni * 16 + c];
  });
}

// ---------------------------------------------------------------------------
// out = LN(a + b) with gain/beta, one 256-wide row per block
// ---------------------------------------------------------------------------
__global__ void k_add_ln(const float* __restrict__ a, const float* __restrict__ b,
                         const float* __restrict__ gain, const float* __restrict__ beta,
                         float* __restrict__ out) {
  __shared__ float sh[Dc];
  const int row = blockIdx.x, c = threadIdx.x;
  const float v = a[(size_t)row * Dc + c] + b[(size_t)row * Dc + c];
  sh[c] = v; __syncthreads();
  for (int o = 128; o > 0; o >>= 1) { if (c < o) sh[c] += sh[c + o]; __syncthreads(); }
  const float m = sh[0] * (1.f / Dc); __syncthreads();
  sh[c] = v * v; __syncthreads();
  for (int o = 128; o > 0; o >>= 1) { if (c < o) sh[c] += sh[c + o]; __syncthreads(); }
  const float var = sh[0] * (1.f / Dc) - m * m;
  out[(size_t)row * Dc + c] = (v - m) * rsqrtf(var + 1e-5f) * gain[c] + beta[c];
}

// ---------------------------------------------------------------------------
// Fused graph branch: LN1 -> FFN (bf16 LDS hidden) -> residual -> LN2 -> store
// 16 rows per block, 8 waves. LDS = 16KB + 32KB + 16KB = 64KB.
// ---------------------------------------------------------------------------
__device__ __forceinline__ void ln_rows(float (*tile)[Dc],
                                        const float* __restrict__ gain,
                                        const float* __restrict__ beta) {
  const int wid = threadIdx.x >> 5, lane = threadIdx.x & 31;
#pragma unroll
  for (int rr = 0; rr < 2; ++rr) {
    const int r = wid * 2 + rr;
    float s = 0.f, s2 = 0.f;
#pragma unroll
    for (int j = 0; j < 8; ++j) { float v = tile[r][lane + 32 * j]; s += v; s2 += v * v; }
    for (int o = 16; o > 0; o >>= 1) { s += __shfl_xor(s, o, 32); s2 += __shfl_xor(s2, o, 32); }
    const float m = s * (1.f / Dc);
    const float inv = rsqrtf(s2 * (1.f / Dc) - m * m + 1e-5f);
#pragma unroll
    for (int j = 0; j < 8; ++j) {
      const int c = lane + 32 * j;
      tile[r][c] = (tile[r][c] - m) * inv * gain[c] + beta[c];
    }
  }
}

__global__ void k_graph_fused(const float* __restrict__ graphs,
                              const float* __restrict__ leftb,
                              const float* __restrict__ rightb,
                              const __bf16* __restrict__ Wgf1t,   // [1024][256]
                              const float* __restrict__ bgf1,
                              const __bf16* __restrict__ Wgf2t,   // [256][1024]
                              const float* __restrict__ bgf2,
                              const float* __restrict__ gg1, const float* __restrict__ betag1,
                              const float* __restrict__ gg2, const float* __restrict__ betag2,
                              float* __restrict__ outg) {
  __shared__ float  sBgn[16][Dc];
  __shared__ __bf16 sHid[16][DFFc];
  __shared__ float  sOut[16][Dc];

  const size_t r0 = (size_t)blockIdx.x * 16;
  const int wid = threadIdx.x >> 5;

  // ---- load graphs + left[x] + right[y] with b128 vectors; prefetch next tile
  {
    const float4* g4 = (const float4*)graphs;
    const float4* l4 = (const float4*)leftb;
    const float4* rr4 = (const float4*)rightb;
#pragma unroll
    for (int j = 0; j < 4; ++j) {
      const int idx = threadIdx.x + 256 * j;          // 0..1023 = 16 rows * 64 float4
      const int r = idx >> 6, c4 = idx & 63;
      const size_t row = r0 + r;
      const int b = (int)(row >> 14);
      const int x = (int)((row >> 7) & 127);
      const int y = (int)(row & 127);
      const float4 g = g4[row * 64 + c4];
      const float4 lv = l4[(size_t)(b * Nn + x) * 64 + c4];
      const float4 rv = rr4[(size_t)(b * Nn + y) * 64 + c4];
      sBgn[r][c4 * 4 + 0] = g.x + lv.x + rv.x;
      sBgn[r][c4 * 4 + 1] = g.y + lv.y + rv.y;
      sBgn[r][c4 * 4 + 2] = g.z + lv.z + rv.z;
      sBgn[r][c4 * 4 + 3] = g.w + lv.w + rv.w;
      if (row + 16 < (size_t)BNN)                     // next tile -> global_prefetch
        __builtin_prefetch(&g4[(row + 16) * 64 + c4], 0, 1);
    }
  }
  __syncthreads();
  ln_rows(sBgn, gg1, betag1);         // bg after first layernorm
  __syncthreads();

  // ---- FFN stage 1: hidden = relu(bgn @ Wgf1 + bgf1), bf16 in LDS
  for (int nt = wid; nt < DFFc / 16; nt += 8) {
    v8f acc = {};
    for (int kb = 0; kb < Dc; kb += 32) {
      v16bf af = frag_a([&](int r, int k) { return sBgn[r][kb + k]; });
      v16bf bf = frag_b([&](int k, int cc) {
        return Wgf1t[(size_t)(nt * 16 + cc) * Dc + kb + k];
      });
      acc = wmma_bf16(acc, af, bf);
    }
    tile_store(acc, [&](int r, int cc, float v) {
      float val = v + bgf1[nt * 16 + cc];
      sHid[r][nt * 16 + cc] = (__bf16)(val > 0.f ? val : 0.f);
    });
  }
  __syncthreads();

  // ---- FFN stage 2 + residual: out = bgn + hidden @ Wgf2 + bgf2
  for (int nt = wid; nt < Dc / 16; nt += 8) {
    v8f acc = {};
    for (int kb = 0; kb < DFFc; kb += 32) {
      v16bf af = frag_a([&](int r, int k) { return sHid[r][kb + k]; });
      v16bf bf = frag_b([&](int k, int cc) {
        return Wgf2t[(size_t)(nt * 16 + cc) * DFFc + kb + k];
      });
      acc = wmma_bf16(acc, af, bf);
    }
    tile_store(acc, [&](int r, int cc, float v) {
      sOut[r][nt * 16 + cc] = v + bgf2[nt * 16 + cc] + sBgn[r][nt * 16 + cc];
    });
  }
  __syncthreads();
  ln_rows(sOut, gg2, betag2);
  __syncthreads();

  // ---- b128 store
  {
    float4* o4 = (float4*)outg;
#pragma unroll
    for (int j = 0; j < 4; ++j) {
      const int idx = threadIdx.x + 256 * j;
      const int r = idx >> 6, c4 = idx & 63;
      o4[(r0 + r) * 64 + c4] = make_float4(sOut[r][c4 * 4 + 0], sOut[r][c4 * 4 + 1],
                                           sOut[r][c4 * 4 + 2], sOut[r][c4 * 4 + 3]);
    }
  }
}

// ---------------------------------------------------------------------------
// launch
// ---------------------------------------------------------------------------
extern "C" void kernel_launch(void* const* d_in, const int* in_sizes, int n_in,
                              void* d_out, int out_size, void* d_ws, size_t ws_size,
                              hipStream_t stream) {
  (void)in_sizes; (void)n_in; (void)out_size; (void)ws_size;

  const float*   x      = (const float*)d_in[0];
  const float*   graphs = (const float*)d_in[1];
  const uint8_t* mask   = (const uint8_t*)d_in[2];
  const float *Wq = (const float*)d_in[3],  *bq = (const float*)d_in[4];
  const float *Wk = (const float*)d_in[5],  *bk = (const float*)d_in[6];
  const float *Wv = (const float*)d_in[7],  *bv = (const float*)d_in[8];
  const float *Wo = (const float*)d_in[9],  *bo = (const float*)d_in[10];
  const float *Wl = (const float*)d_in[11], *bl = (const float*)d_in[12];
  const float *Wr = (const float*)d_in[13], *br = (const float*)d_in[14];
  const float *Wrk = (const float*)d_in[15];
  const float *Wrv = (const float*)d_in[17], *brv = (const float*)d_in[18];
  const float *W1 = (const float*)d_in[19], *b1 = (const float*)d_in[20];
  const float *W2 = (const float*)d_in[21], *b2 = (const float*)d_in[22];
  const float *Wgf1 = (const float*)d_in[23], *bgf1 = (const float*)d_in[24];
  const float *Wgf2 = (const float*)d_in[25], *bgf2 = (const float*)d_in[26];
  const float *g1 = (const float*)d_in[27], *g2 = (const float*)d_in[28];
  const float *gg1 = (const float*)d_in[29], *gg2 = (const float*)d_in[30];
  const float *beta1 = (const float*)d_in[31], *beta2 = (const float*)d_in[32];
  const float *betag1 = (const float*)d_in[33], *betag2 = (const float*)d_in[34];

  float* ws = (float*)d_ws;
  // ---- f32 workspace (~16.5 MiB)
  float* q    = ws;                         // 512*256
  float* kf   = q    + BN * Dc;
  float* vf   = kf   + BN * Dc;
  float* R    = vf   + BN * Dc;             // 512*8*256
  float* S    = R    + BN * Hc * Dc;        // 512*128*8
  float* T    = S    + BN * Nn * Hc;        // 512*8*256
  float* attn = T    + BN * Hc * Dc;
  float* x2   = attn + BN * Dc;
  float* lft  = x2   + BN * Dc;
  float* rgt  = lft  + BN * Dc;
  float* xres = rgt  + BN * Dc;
  float* h1   = xres + BN * Dc;             // 512*1024
  float* xff  = h1   + BN * DFFc;
  float* fend = xff  + BN * Dc;

  // ---- bf16 transposed-weight workspace (~2.9 MiB)
  __bf16* wb   = (__bf16*)fend;
  __bf16* Wqt  = wb;                        // [256][256]
  __bf16* Wkt  = Wqt  + Dc * Dc;
  __bf16* Wvt  = Wkt  + Dc * Dc;
  __bf16* Wot  = Wvt  + Dc * Dc;
  __bf16* Wlt  = Wot  + Dc * Dc;
  __bf16* Wrt  = Wlt  + Dc * Dc;
  __bf16* Wrvt = Wrt  + Dc * Dc;
  __bf16* W1t  = Wrvt + Dc * Dc;            // [1024][256]
  __bf16* W2t  = W1t  + Dc * DFFc;          // [256][1024]
  __bf16* Wgf1t = W2t  + DFFc * Dc;         // [1024][256]
  __bf16* Wgf2t = Wgf1t + Dc * DFFc;        // [256][1024]

  float* out_x  = (float*)d_out;            // (B,N,D)
  float* out_bg = out_x + BN * Dc;          // (B,N,N,D)

  const int waves = 4, blk = waves * 32;
  auto tiles = [](int M, int Nc2) { return (M / 16) * (Nc2 / 16); };
  auto gdim  = [&](int t) { return (t + waves - 1) / waves; };

  // 0. weight transpose+convert to bf16 [N][K]
  k_wt_bf16<<<dim3(Dc/16,  Dc/16),  256, 0, stream>>>(Wq,   Wqt,   Dc,   Dc);
  k_wt_bf16<<<dim3(Dc/16,  Dc/16),  256, 0, stream>>>(Wk,   Wkt,   Dc,   Dc);
  k_wt_bf16<<<dim3(Dc/16,  Dc/16),  256, 0, stream>>>(Wv,   Wvt,   Dc,   Dc);
  k_wt_bf16<<<dim3(Dc/16,  Dc/16),  256, 0, stream>>>(Wo,   Wot,   Dc,   Dc);
  k_wt_bf16<<<dim3(Dc/16,  Dc/16),  256, 0, stream>>>(Wl,   Wlt,   Dc,   Dc);
  k_wt_bf16<<<dim3(Dc/16,  Dc/16),  256, 0, stream>>>(Wr,   Wrt,   Dc,   Dc);
  k_wt_bf16<<<dim3(Dc/16,  Dc/16),  256, 0, stream>>>(Wrv,  Wrvt,  Dc,   Dc);
  k_wt_bf16<<<dim3(Dc/16,  DFFc/16),256, 0, stream>>>(W1,   W1t,   Dc,   DFFc);
  k_wt_bf16<<<dim3(DFFc/16,Dc/16),  256, 0, stream>>>(W2,   W2t,   DFFc, Dc);
  k_wt_bf16<<<dim3(Dc/16,  DFFc/16),256, 0, stream>>>(Wgf1, Wgf1t, Dc,   DFFc);
  k_wt_bf16<<<dim3(DFFc/16,Dc/16),  256, 0, stream>>>(Wgf2, Wgf2t, DFFc, Dc);

  // 1. node projections q, k, v
  k_gemm_bias<<<gdim(tiles(BN, Dc)), blk, 0, stream>>>(x, Dc, Wqt, bq, q,  Dc, BN, Dc, Dc, 0);
  k_gemm_bias<<<gdim(tiles(BN, Dc)), blk, 0, stream>>>(x, Dc, Wkt, bk, kf, Dc, BN, Dc, Dc, 0);
  k_gemm_bias<<<gdim(tiles(BN, Dc)), blk, 0, stream>>>(x, Dc, Wvt, bv, vf, Dc, BN, Dc, Dc, 0);

  // 2. R = per-head Wrk^T q (replaces the gk tensor)
  k_relR<<<dim3(Dc / 16, BN / 16, Hc), 32, 0, stream>>>(q, Wrk, R);

  // 3. scores + softmax
  k_scores<<<dim3(Nn / 16, BN), 32, 0, stream>>>(graphs, R, q, kf, mask, S);
  k_softmax<<<BN * Hc, Nn, 0, stream>>>(S);

  // 4. t = p^T graphs (replaces the gv tensor), then attn = p@v + t@Wrv + brv
  k_t<<<dim3(Dc / 16, BN), 32, 0, stream>>>(S, graphs, T);
  k_attn<<<dim3(DKc / 16, Nn / 16, Bc * Hc), 32, 0, stream>>>(S, vf, T, Wrvt, brv, attn);

  // 5. output projections
  k_gemm_bias<<<gdim(tiles(BN, Dc)), blk, 0, stream>>>(attn, Dc, Wot, bo, x2,  Dc, BN, Dc, Dc, 0);
  k_gemm_bias<<<gdim(tiles(BN, Dc)), blk, 0, stream>>>(attn, Dc, Wlt, bl, lft, Dc, BN, Dc, Dc, 0);
  k_gemm_bias<<<gdim(tiles(BN, Dc)), blk, 0, stream>>>(attn, Dc, Wrt, br, rgt, Dc, BN, Dc, Dc, 0);

  // 6. node branch: LN -> FFN -> LN
  k_add_ln<<<BN, Dc, 0, stream>>>(x, x2, g1, beta1, xres);
  k_gemm_bias<<<gdim(tiles(BN, DFFc)), blk, 0, stream>>>(xres, Dc, W1t, b1, h1, DFFc, BN, DFFc, Dc, 1);
  k_gemm_bias<<<gdim(tiles(BN, Dc)),  blk, 0, stream>>>(h1, DFFc, W2t, b2, xff, Dc, BN, Dc, DFFc, 0);
  k_add_ln<<<BN, Dc, 0, stream>>>(xres, xff, g2, beta2, out_x);

  // 7. graph branch, fully fused (LN1 -> FFN -> LN2), 16 rows/block
  k_graph_fused<<<(int)(BNN / 16), 256, 0, stream>>>(
      graphs, lft, rgt, Wgf1t, bgf1, Wgf2t, bgf2, gg1, betag1, gg2, betag2, out_bg);
}
